// GRUEncoder_89730456748115
// MI455X (gfx1250) — matrix operand order repaired and checked by reading, not compiled, not run on hardware
//
#include <hip/hip_runtime.h>
#include <hip/hip_bf16.h>
#include <stdint.h>

// CDNA5 / gfx1250: wave32, WMMA 16x16x32 bf16, 320KB LDS per WGP.

typedef __attribute__((ext_vector_type(16))) __bf16 v16bf;
typedef __attribute__((ext_vector_type(8)))  float  v8f;

#define BATCH 1024
#define SEQ   200
#define IN0   64
#define HID   128
#define G3    384   // 3*HID

union Frag {
    uint4 q[2];     // 32 bytes = 16 bf16
    v16bf v;
};

__device__ __forceinline__ unsigned short f2bf(float f) {
    uint32_t u = __float_as_uint(f);
    u += 0x7FFFu + ((u >> 16) & 1u);    // round-to-nearest-even
    return (unsigned short)(u >> 16);
}

__device__ __forceinline__ v8f splat8(float f) {
    v8f r;
#pragma unroll
    for (int i = 0; i < 8; ++i) r[i] = f;
    return r;
}

// Load A-matrix fragment (16xK bf16 tile, row-major in LDS) for K-chunk kc.
// ISA layout: lane = 16*half + m ; VGPR0-3 hold K = 32kc + 8*half + [0..7],
// VGPR4-7 hold K = 32kc + 16 + 8*half + [0..7]  -> two b128 LDS loads.
template<int K, int NK>
__device__ __forceinline__ void load_A(const unsigned short* src, int lane, Frag* a) {
    const int half = lane >> 4;
    const unsigned short* row = src + (lane & 15) * K;
#pragma unroll
    for (int kc = 0; kc < NK; ++kc) {
        a[kc].q[0] = *(const uint4*)(row + kc * 32 + 8 * half);
        a[kc].q[1] = *(const uint4*)(row + kc * 32 + 16 + 8 * half);
    }
}

// Accumulate acc += A(16xK) * W[row0..row0+16)(^T as B) over NK K-chunks.
// B layout: lane = 16*half + n ; VGPR0-3 hold K = 32kc + 8*half + [0..7],
// VGPR4-7 hold K = 32kc + 16 + 8*half + [0..7] -> two b128 LDS loads.
template<int K, int NK>
__device__ __forceinline__ v8f mm_row(const unsigned short* __restrict__ W,
                                      int row0, int lane, const Frag* a, v8f acc) {
    const int half = lane >> 4;
    const unsigned short* wrow = W + (size_t)(row0 + (lane & 15)) * K + 8 * half;
#pragma unroll
    for (int kc = 0; kc < NK; ++kc) {
        Frag b;
        const uint4* p = (const uint4*)(wrow + kc * 32);
        b.q[0] = p[0];
        b.q[1] = p[2];
        acc = __builtin_amdgcn_wmma_f32_16x16x32_bf16(
            /*neg_a=*/false, a[kc].v, /*neg_b=*/false, b.v,
            /*c_mod=*/(short)0, acc, /*reuse_a=*/false, /*reuse_b=*/false);
    }
    return acc;
}

// One GRU layer: each block owns a 16-row batch tile for all SEQ steps.
// 8 waves; wave w produces hidden columns [16w, 16w+16) of r/z/n and h.
template<int KIN, bool IN_F32, bool OUT_SEQ>
__global__ __launch_bounds__(256) void gru_layer_kernel(
    const void*  __restrict__ Xv,         // (B, SEQ, KIN)  f32 or bf16
    const float* __restrict__ W_ih,       // (384, KIN) f32
    const float* __restrict__ W_hh,       // (384, 128) f32
    const float* __restrict__ b_ih,       // (384,)
    const float* __restrict__ b_hh,       // (384,)
    unsigned short* __restrict__ out_seq, // (B, SEQ, 128) bf16   (layer 0)
    float*          __restrict__ out_last // (B, 128) f32         (layer 1)
) {
    __shared__ __align__(16) unsigned short w_ih_lds[G3 * KIN];
    __shared__ __align__(16) unsigned short w_hh_lds[G3 * HID];
    __shared__ __align__(16) unsigned short h_lds[16 * HID];
    __shared__ __align__(16) unsigned short x_lds[16 * KIN];

    const int tid  = threadIdx.x;
    const int wv   = tid >> 5;
    const int lane = tid & 31;
    const int half = lane >> 4;
    const int b0   = blockIdx.x * 16;
    const int hc   = wv * 16 + (lane & 15);   // hidden column this lane owns

    // Stage weights (f32 -> bf16, row-major) and zero h.
    for (int i = tid; i < G3 * KIN; i += 256) w_ih_lds[i] = f2bf(W_ih[i]);
    for (int i = tid; i < G3 * HID; i += 256) w_hh_lds[i] = f2bf(W_hh[i]);
    for (int i = tid; i < 16 * HID; i += 256) h_lds[i] = 0;

    const float br  = b_ih[hc]           + b_hh[hc];
    const float bz  = b_ih[HID + hc]     + b_hh[HID + hc];
    const float bxn = b_ih[2 * HID + hc];
    const float bhn = b_hh[2 * HID + hc];

    v8f h = splat8(0.0f);                 // C/D fragment: this wave's h slice
    constexpr int NKX = KIN / 32;
    Frag ah[4], ax[NKX];

    const int sr = tid >> 4;              // staging row 0..15
    const int sc = tid & 15;

    for (int t = 0; t < SEQ; ++t) {
        // ---- stage x_t tile (16 x KIN) into LDS as bf16 ----
        if constexpr (IN_F32) {
            const float* X = (const float*)Xv;
            const float* src = X + ((size_t)(b0 + sr) * SEQ + t) * KIN + sc * 4;
            float4 v4 = *(const float4*)src;
            unsigned short* d = x_lds + sr * KIN + sc * 4;
            d[0] = f2bf(v4.x); d[1] = f2bf(v4.y); d[2] = f2bf(v4.z); d[3] = f2bf(v4.w);
            if (t + 1 < SEQ) __builtin_prefetch(src + KIN, 0, 0);
        } else {
            const unsigned short* X = (const unsigned short*)Xv;
            const unsigned short* src = X + ((size_t)(b0 + sr) * SEQ + t) * KIN + sc * 8;
            *(uint4*)(x_lds + sr * KIN + sc * 8) = *(const uint4*)src;
            if (t + 1 < SEQ) __builtin_prefetch(src + KIN, 0, 0);
        }
        __syncthreads();   // x_lds ready; h_lds from t-1 (or init) visible

        // ---- gate GEMMs: 16 x 384 x (KIN + HID), bf16 WMMA ----
        load_A<HID, 4>(h_lds, lane, ah);
        load_A<KIN, NKX>(x_lds, lane, ax);

        v8f ar = mm_row<HID, 4>(w_hh_lds, 16 * wv, lane, ah, splat8(br));
        ar     = mm_row<KIN, NKX>(w_ih_lds, 16 * wv, lane, ax, ar);
        v8f az = mm_row<HID, 4>(w_hh_lds, HID + 16 * wv, lane, ah, splat8(bz));
        az     = mm_row<KIN, NKX>(w_ih_lds, HID + 16 * wv, lane, ax, az);
        v8f an_h = mm_row<HID, 4>(w_hh_lds, 2 * HID + 16 * wv, lane, ah, splat8(bhn));
        v8f an_x = mm_row<KIN, NKX>(w_ih_lds, 2 * HID + 16 * wv, lane, ax, splat8(bxn));

        // ---- gate nonlinearities + state update (wave-local) ----
#pragma unroll
        for (int i = 0; i < 8; ++i) {
            float r    = 1.0f / (1.0f + __expf(-ar[i]));
            float z    = 1.0f / (1.0f + __expf(-az[i]));
            float npre = an_x[i] + r * an_h[i];
            float n    = 2.0f / (1.0f + __expf(-2.0f * npre)) - 1.0f;  // tanh
            h[i] = n + z * (h[i] - n);
        }

        __syncthreads();   // everyone done reading h_lds/x_lds of this step

        // ---- publish h (bf16) for next step's A matrix; emit sequence ----
#pragma unroll
        for (int i = 0; i < 8; ++i) {
            int m = i + 8 * half;                 // C/D layout: M = i + 8*(lane/16)
            unsigned short hb = f2bf(h[i]);
            h_lds[m * HID + hc] = hb;
            if constexpr (OUT_SEQ)
                out_seq[((size_t)(b0 + m) * SEQ + t) * HID + hc] = hb;
        }
    }

    if constexpr (!OUT_SEQ) {
#pragma unroll
        for (int i = 0; i < 8; ++i) {
            int m = i + 8 * half;
            out_last[(size_t)(b0 + m) * HID + hc] = h[i];
        }
    }
}

extern "C" void kernel_launch(void* const* d_in, const int* in_sizes, int n_in,
                              void* d_out, int out_size, void* d_ws, size_t ws_size,
                              hipStream_t stream) {
    const float* x     = (const float*)d_in[0];
    const float* W_ih0 = (const float*)d_in[1];
    const float* W_hh0 = (const float*)d_in[2];
    const float* b_ih0 = (const float*)d_in[3];
    const float* b_hh0 = (const float*)d_in[4];
    const float* W_ih1 = (const float*)d_in[5];
    const float* W_hh1 = (const float*)d_in[6];
    const float* b_ih1 = (const float*)d_in[7];
    const float* b_hh1 = (const float*)d_in[8];

    // Workspace: layer-0 output sequence, bf16 (B, SEQ, HID) = 50 MB.
    unsigned short* h1seq = (unsigned short*)d_ws;

    dim3 grid(BATCH / 16), block(256);
    gru_layer_kernel<IN0, true, true><<<grid, block, 0, stream>>>(
        (const void*)x, W_ih0, W_hh0, b_ih0, b_hh0, h1seq, nullptr);
    gru_layer_kernel<HID, false, false><<<grid, block, 0, stream>>>(
        (const void*)h1seq, W_ih1, W_hh1, b_ih1, b_hh1, nullptr, (float*)d_out);
}
